// SimGCL_Encoder_12721693131117
// MI455X (gfx1250) — compile-verified, matrix operand order checked by reading, not run on hardware
//
#include <hip/hip_runtime.h>
#include <stdint.h>

// SimGCL encoder for MI455X (gfx1250, wave32).
//
// Per layer k (3 layers):
//   1) ego_next = SpMM(COO edges, ego)     -- f32 atomic scatter-add, L2-resident
//   2) ego_next += sign(ego_next) * (noise_row / ||noise_row||2) * 0.1
//      acc      += ego_next                -- acc lives in d_out; last layer scales by 1/3
//
// Roofline: ~2.3 GB of traffic, working set ~77 MB << 192 MB L2 => L2-bandwidth /
// atomic-throughput bound. FLOPs trivial. The v1 noise kernel was VALU-bound on the
// hash PRNG (valu=1046 vs 32 vmem); v2 generates two 16-bit uniforms per hash and
// shares the hash between the WMMA-norm phase and the update phase (3x fewer VALU).
//
// CDNA5 paths: v_wmma_f32_16x16x32_f16 row-sums (A = noise^2 in ISA A-layout,
// B = ones, f32 accum), global_prefetch_b8 edge-stream prefetch, single-instruction
// global_atomic_add_f32 no-return scatter (verified: no CAS loop in histogram).
//
// Workspace: needs 2 * N * EMB * 4 = 51.2 MB in d_ws (ping-pong ego buffers).

#define N_USERS 70000
#define N_ITEMS 30000
#define NN      (N_USERS + N_ITEMS)   // 100000 rows
#define NE      1280000               // edges
#define EMB     64
#define NLAYERS 3
#define EPSV    0.1f

typedef __attribute__((ext_vector_type(16))) _Float16 v16h;
typedef __attribute__((ext_vector_type(8)))  float    v8f;

// One 32-bit hash -> two uniform [0,1) values (16-bit resolution each; noise
// quality only -- no bit-exactness requirement in this compile-only setting).
// pidx identifies an even-aligned column pair: pidx = row*32 + col/2.
__device__ __forceinline__ float2 urand_pair(uint32_t layer, uint32_t pidx) {
  uint32_t x = (pidx ^ (layer * 0x85EBCA6Bu) ^ 42u) * 0x9E3779B9u;
  x ^= x >> 16;
  x *= 0x7FEB352Du;
  x ^= x >> 15;
  float2 r;
  r.x = (float)(x & 0xFFFFu) * 0x1p-16f;
  r.y = (float)(x >> 16)     * 0x1p-16f;
  return r;
}

// sign(e) * t  with sign(0) == 0.
__device__ __forceinline__ float signed_mag(float e, float t) {
  return (e != 0.f) ? __builtin_copysignf(t, e) : 0.f;
}

// ---------------------------------------------------------------------------
// SpMM: out[row,:] += val * x[col,:].  16 lanes per edge, float4 per lane
// => one half-wave moves a full 256B embedding row, fully coalesced.
// Scatter via fire-and-forget f32 atomics (no return => STOREcnt only).
// ---------------------------------------------------------------------------
__global__ __launch_bounds__(256) void spmm_kernel(
    const int*   __restrict__ erow,
    const int*   __restrict__ ecol,
    const float* __restrict__ eval,
    const float* __restrict__ x,
    float*       __restrict__ out) {
  uint32_t t   = blockIdx.x * 256u + threadIdx.x;
  uint32_t e   = t >> 4;           // edge index (16 lanes per edge)
  if (e >= NE) return;
  uint32_t sub = t & 15u;          // which float4 of the 64-wide row

  // Prefetch edge streams well ahead (linear). Speculative; OOB is dropped.
  __builtin_prefetch(ecol + e + 4096, 0, 1);
  __builtin_prefetch(erow + e + 4096, 0, 1);
  __builtin_prefetch(eval + e + 4096, 0, 1);

  int   r = erow[e];
  int   c = ecol[e];
  float v = eval[e];

  const float4 xv = *(const float4*)(x + (size_t)c * EMB + sub * 4u);
  float* dst = out + (size_t)r * EMB + sub * 4u;
  atomicAdd(dst + 0, v * xv.x);
  atomicAdd(dst + 1, v * xv.y);
  atomicAdd(dst + 2, v * xv.z);
  atomicAdd(dst + 3, v * xv.w);
}

// ---------------------------------------------------------------------------
// Noise + accumulate. One wave handles 16 rows.
// Phase 1: A = noise^2 in ISA 16-bit A-matrix (16x32) layout; two chained
//          v_wmma_f32_16x16x32_f16 with B = ones => 16 row-sums in D (f32).
// Phase 2: broadcast rowsums from the documented C/D layout, regenerate the
//          same uniforms in coalesced layout, update ego and acc.
// Grid sized so every wave owns exactly 16 valid rows => EXEC all-ones (WMMA req).
// ---------------------------------------------------------------------------
__global__ __launch_bounds__(320) void noise_acc_kernel(
    float* __restrict__ ego,      // SpMM output of this layer (updated in place)
    float* __restrict__ acc,      // accumulator (= d_out)
    uint32_t layer,
    float final_scale) {          // 1.0 for layers 0..1, 1/3 for last layer
  const uint32_t lane = threadIdx.x & 31u;
  const uint32_t wave = (blockIdx.x * 320u + threadIdx.x) >> 5;
  const uint32_t r0   = wave * 16u;        // first of this wave's 16 rows
  const uint32_t m    = lane & 15u;        // row within tile (A-matrix M)
  const uint32_t h    = lane >> 4;         // half-wave selector

  // ---- Phase 1: WMMA row-sums of squared noise --------------------------
  v8f d = {0.f, 0.f, 0.f, 0.f, 0.f, 0.f, 0.f, 0.f};
  v16h bones;
#pragma unroll
  for (int i = 0; i < 16; ++i) bones[i] = (_Float16)1.0f;

  const uint32_t rowpair_base = (r0 + m) * 32u;   // 32 column-pairs per row

#pragma unroll
  for (int t = 0; t < 2; ++t) {            // K-chunks: cols [0,32) and [32,64)
    v16h a;
#pragma unroll
    for (int j = 0; j < 8; ++j) {
      // 16-bit A-matrix 16x32 layout (ISA 7.12.2): VGPR j holds K pair
      //   j<4 : K = 2j + 8h ; j>=4 : K = 16 + 2(j-4) + 8h   (K always even)
      int K = (j < 4) ? (2 * j + 8 * (int)h) : (16 + 2 * (j - 4) + 8 * (int)h);
      uint32_t col = 32u * (uint32_t)t + (uint32_t)K;
      float2 u = urand_pair(layer, rowpair_base + (col >> 1));
      a[2 * j]     = (_Float16)(u.x * u.x);
      a[2 * j + 1] = (_Float16)(u.y * u.y);
    }
    // D = A*ones + D : every column of D holds the row-sum of u^2.
    d = __builtin_amdgcn_wmma_f32_16x16x32_f16(
        /*neg_a=*/false, a, /*neg_b=*/false, bones,
        /*c_mod=*/(short)0, d, /*reuse_a=*/false, /*reuse_b=*/false);
  }

  // ---- Phase 2: apply noise, accumulate ---------------------------------
  // C/D layout: VGPR v holds M=v (lanes 0-15) and M=v+8 (lanes 16-31).
#pragma unroll
  for (int p = 0; p < 8; ++p) {
    const int mA = 2 * p;       // row handled by lanes 0-15 this iteration
    const int mB = 2 * p + 1;   // row handled by lanes 16-31
    float sA = __shfl(d[mA & 7], (mA >= 8) ? 16 : 0, 32);
    float sB = __shfl(d[mB & 7], (mB >= 8) ? 16 : 0, 32);
    float s  = h ? sB : sA;
    // eps / max(||noise||, 1e-12)  ==  eps * rsqrt(max(s, 1e-24))
    float inv = EPSV * rsqrtf(fmaxf(s, 1e-24f));

    uint32_t row = r0 + (uint32_t)(2 * p) + h;
    size_t   off = (size_t)row * EMB + m * 4u;

    // Same pair-hashes as phase 1: cols [4m,4m+3] = pairs 2m and 2m+1.
    uint32_t pbase = row * 32u + m * 2u;
    float2 ua = urand_pair(layer, pbase);
    float2 ub = urand_pair(layer, pbase + 1u);

    float4 e = *(const float4*)(ego + off);
    e.x += signed_mag(e.x, ua.x * inv);
    e.y += signed_mag(e.y, ua.y * inv);
    e.z += signed_mag(e.z, ub.x * inv);
    e.w += signed_mag(e.w, ub.y * inv);
    *(float4*)(ego + off) = e;

    float4 a4 = *(const float4*)(acc + off);
    a4.x = (a4.x + e.x) * final_scale;
    a4.y = (a4.y + e.y) * final_scale;
    a4.z = (a4.z + e.z) * final_scale;
    a4.w = (a4.w + e.w) * final_scale;
    *(float4*)(acc + off) = a4;
  }
}

// ---------------------------------------------------------------------------
extern "C" void kernel_launch(void* const* d_in, const int* in_sizes, int n_in,
                              void* d_out, int out_size, void* d_ws, size_t ws_size,
                              hipStream_t stream) {
  const float* ego0 = (const float*)d_in[0];   // (N, 64) fp32
  const int*   erow = (const int*)  d_in[1];   // (E,) int32
  const int*   ecol = (const int*)  d_in[2];   // (E,) int32
  const float* eval = (const float*)d_in[3];   // (E,) fp32
  float* out = (float*)d_out;                  // (N, 64) fp32 == acc buffer

  const size_t tab_elems = (size_t)NN * EMB;   // 6.4M floats = 25.6 MB
  float* bufA = (float*)d_ws;                  // needs 2*25.6 MB of workspace
  float* bufB = bufA + tab_elems;

  // acc = 0 (d_out is poisoned by the harness; must re-init every call)
  hipMemsetAsync(out, 0, tab_elems * sizeof(float), stream);

  const float* x   = ego0;   // layer 0 reads the original embeddings
  float*       cur = bufA;

  const uint32_t spmm_blocks  = (NE * 16u + 255u) / 256u;  // 80000
  const uint32_t noise_blocks = (NN / 16u) / 10u;          // 6250 waves / 10 per block = 625

  for (int k = 0; k < NLAYERS; ++k) {
    hipMemsetAsync(cur, 0, tab_elems * sizeof(float), stream);
    spmm_kernel<<<dim3(spmm_blocks), dim3(256), 0, stream>>>(erow, ecol, eval, x, cur);
    const float fs = (k == NLAYERS - 1) ? (1.0f / 3.0f) : 1.0f;
    noise_acc_kernel<<<dim3(noise_blocks), dim3(320), 0, stream>>>(cur, out, (uint32_t)k, fs);
    x   = cur;
    cur = (cur == bufA) ? bufB : bufA;
  }
  // out = (sum of 3 layer embeddings) / 3, rows [0,70000) users then [70000,100000) items,
  // which is exactly the concatenated tuple layout the harness expects.
}
